// DecoderBlock_76527727280436
// MI455X (gfx1250) — matrix operand order repaired and checked by reading, not compiled, not run on hardware
//
#include <hip/hip_runtime.h>
#include <cmath>

typedef __attribute__((ext_vector_type(8)))  float  v8f;
typedef __attribute__((ext_vector_type(16))) __bf16 v16bf;
typedef int v4i_vs __attribute__((vector_size(16)));  // matches builtin param type

#define kB 2
#define kT 2048
#define kC 1024
#define kH 16
#define kD 64
#define kF 4096
#define kR (kB * kT)

// gfx1250 async global->LDS path (ASYNCcnt-tracked); guarded so absence of the
// builtins degrades to plain b128 copies instead of breaking the build.
#if __has_builtin(__builtin_amdgcn_global_load_async_to_lds_b128) && \
    __has_builtin(__builtin_amdgcn_s_wait_asynccnt)
#define USE_ASYNC_LDS 1
#else
#define USE_ASYNC_LDS 0
#endif

__device__ __forceinline__ void copy_b128_to_lds(const void* g, void* l) {
#if USE_ASYNC_LDS
  __builtin_amdgcn_global_load_async_to_lds_b128(
      (__attribute__((address_space(1))) v4i_vs*)(void*)g,
      (__attribute__((address_space(3))) v4i_vs*)l, 0, 0);
#else
  *reinterpret_cast<uint4*>(l) = *reinterpret_cast<const uint4*>(g);
#endif
}

__device__ __forceinline__ void wait_async_lds() {
#if USE_ASYNC_LDS
  __builtin_amdgcn_s_wait_asynccnt(0);
#endif
}

// ---------------------------------------------------------------- cast f32 -> bf16
__global__ void cast_f32_to_bf16(const float* __restrict__ in, __bf16* __restrict__ out, int n) {
  int i = blockIdx.x * blockDim.x + threadIdx.x;
  int stride = gridDim.x * blockDim.x;
  for (; i < n; i += stride) out[i] = (__bf16)in[i];
}

// ---------------------------------------------------------------- layernorm (row per block) -> bf16
__global__ void __launch_bounds__(256) layernorm_bf16(const float* __restrict__ x,
                                                      const float* __restrict__ g,
                                                      const float* __restrict__ beta,
                                                      __bf16* __restrict__ out) {
  __shared__ float sred[16];
  const int row = blockIdx.x;
  const float* xr = x + (size_t)row * kC;
  float s = 0.f, s2 = 0.f;
#pragma unroll
  for (int e = 0; e < kC / 256; ++e) {
    const float v = xr[threadIdx.x + e * 256];
    s += v;
    s2 += v * v;
  }
#pragma unroll
  for (int off = 16; off > 0; off >>= 1) {
    s += __shfl_xor(s, off, 32);
    s2 += __shfl_xor(s2, off, 32);
  }
  const int wid = threadIdx.x >> 5;
  if ((threadIdx.x & 31) == 0) {
    sred[wid] = s;
    sred[8 + wid] = s2;
  }
  __syncthreads();
  s = 0.f;
  s2 = 0.f;
#pragma unroll
  for (int w = 0; w < 8; ++w) {
    s += sred[w];
    s2 += sred[8 + w];
  }
  const float mu = s * (1.0f / kC);
  const float var = s2 * (1.0f / kC) - mu * mu;
  const float rstd = rsqrtf(var + 1e-3f);
  __bf16* outr = out + (size_t)row * kC;
#pragma unroll
  for (int e = 0; e < kC / 256; ++e) {
    const int c = threadIdx.x + e * 256;
    outr[c] = (__bf16)((xr[c] - mu) * rstd * g[c] + beta[c]);
  }
}

// ---------------------------------------------------------------- WMMA GEMM: out = A[MxK](bf16) * W[KxN](bf16) + bias
// Double-buffered LDS, async global->LDS staging for A, one barrier per K-step.
// EPI 0: store bf16(v) ; EPI 1: store bf16(gelu_exact(v)) ; EPI 2: store f32(v + res)
template <int EPI>
__global__ void __launch_bounds__(256) gemm_bf16_wmma(const __bf16* __restrict__ A,
                                                      const __bf16* __restrict__ W,
                                                      const float* __restrict__ bias,
                                                      const float* __restrict__ res,
                                                      __bf16* __restrict__ outb,
                                                      float* __restrict__ outf,
                                                      int M, int N, int K) {
  constexpr int TM = 128, TN = 128, TK = 32;
  __shared__ __bf16 As[2][TM * TK];  // [m][k]
  __shared__ __bf16 Bs[2][TN * TK];  // [n][k]  (transposed for contiguous B-fragment loads)
  const int bm = blockIdx.x * TM, bn = blockIdx.y * TN;
  const int tid = threadIdx.x, lane = tid & 31, wid = tid >> 5;
  const int half = lane >> 4, l15 = lane & 15;
  const int wm = (wid & 1) * 64, wn = (wid >> 1) * 32;

  const int ar = tid >> 1, ako = (tid & 1) * 16;  // A staging coords (16 bf16 / thread)
  const int bk = tid >> 3, bno = (tid & 7) * 16;  // B staging coords (16 bf16 / thread)

  auto stage = [&](int k0, int buf) {
    // A tile: two async b128 per thread, contiguous
    const __bf16* ga = A + (size_t)(bm + ar) * K + k0 + ako;
    __bf16* la = &As[buf][ar * TK + ako];
    copy_b128_to_lds(ga, la);
    copy_b128_to_lds(ga + 8, la + 8);
    // B tile: plain b128 loads, scatter transposed into LDS
    const uint4* s4 = reinterpret_cast<const uint4*>(W + (size_t)(k0 + bk) * N + bn + bno);
    uint4 b0 = s4[0], b1 = s4[1];
    if (k0 + TK < K) __builtin_prefetch(s4 + (size_t)TK * N / 8, 0, 0);
    __bf16 tmp[16];
    *reinterpret_cast<uint4*>(&tmp[0]) = b0;
    *reinterpret_cast<uint4*>(&tmp[8]) = b1;
#pragma unroll
    for (int j = 0; j < 16; ++j) Bs[buf][(bno + j) * TK + bk] = tmp[j];
  };

  v8f acc[4][2] = {};
  const int nk = K / TK;
  stage(0, 0);
  for (int t = 0; t < nk; ++t) {
    wait_async_lds();
    __syncthreads();
    if (t + 1 < nk) stage((t + 1) * TK, (t + 1) & 1);  // prefetch next tile under the WMMAs

    const __bf16* Ab = As[t & 1];
    const __bf16* Bb = Bs[t & 1];
    v16bf af[4], bfr[2];
#pragma unroll
    for (int i = 0; i < 4; ++i) {  // A 16x32 bf16: e -> k = half*16+e
      const __bf16* p = &Ab[(wm + i * 16 + l15) * TK + half * 16];
#pragma unroll
      for (int e = 0; e < 8; ++e) {
        af[i][e] = p[e];
        af[i][8 + e] = p[8 + e];
      }
    }
#pragma unroll
    for (int j = 0; j < 2; ++j) {  // B 32x16 bf16: lane=col, e -> k = half*16+e
      const __bf16* p = &Bb[(wn + j * 16 + l15) * TK + half * 16];
#pragma unroll
      for (int e = 0; e < 16; ++e) bfr[j][e] = p[e];
    }
#pragma unroll
    for (int i = 0; i < 4; ++i)
#pragma unroll
      for (int j = 0; j < 2; ++j)
        acc[i][j] = __builtin_amdgcn_wmma_f32_16x16x32_bf16(false, af[i], false, bfr[j],
                                                            (short)0, acc[i][j], false, false);
  }

  // epilogue — C/D layout: VGPR m, lanes 0-15 row m, lanes 16-31 row m+8; col = lane&15
#pragma unroll
  for (int i = 0; i < 4; ++i)
#pragma unroll
    for (int j = 0; j < 2; ++j)
#pragma unroll
      for (int m = 0; m < 8; ++m) {
        const int row = bm + wm + i * 16 + m + half * 8;
        const int col = bn + wn + j * 16 + l15;
        const size_t idx = (size_t)row * N + col;
        float v = acc[i][j][m] + bias[col];
        if (EPI == 0) {
          outb[idx] = (__bf16)v;
        } else if (EPI == 1) {
          outb[idx] = (__bf16)(0.5f * v * (1.0f + erff(v * 0.70710678118f)));
        } else {
          outf[idx] = v + res[idx];
        }
      }
}

// ---------------------------------------------------------------- flash attention (causal, scores = K·Qᵀ)
__global__ void __launch_bounds__(128) attn_flash(const __bf16* __restrict__ qb,
                                                  const __bf16* __restrict__ kb,
                                                  const __bf16* __restrict__ vb,
                                                  const float* __restrict__ x,
                                                  float* __restrict__ x1) {
  __shared__ __bf16 Qs[64 * 64];     // [j][d]  (natural layout == transposed B storage)
  __shared__ __bf16 Vs[64 * 64];     // [d][j]  transposed
  __shared__ __bf16 Ps[4][16 * 64];  // per-wave P (C-layout -> A-layout staging)
  const int ti = blockIdx.x, h = blockIdx.y, b = blockIdx.z;
  const int tid = threadIdx.x, lane = tid & 31, wid = tid >> 5;
  const int half = lane >> 4, l15 = lane & 15;
  const int i0 = ti * 64 + wid * 16;  // this wave's 16 "i" rows

  // K A-fragments for rows i0..i0+15 over d = 0..63 (two 16x32 slices)
  v16bf kf[2];
  {
    const __bf16* kr = kb + (size_t)(b * kT + i0 + l15) * kC + h * kD;
#pragma unroll
    for (int f = 0; f < 2; ++f)
#pragma unroll
      for (int e = 0; e < 8; ++e) {
        kf[f][e] = kr[f * 32 + half * 16 + e];
        kf[f][8 + e] = kr[f * 32 + half * 16 + 8 + e];
      }
  }

  float rmax[8], rsum[8];
  v8f o[4] = {};
#pragma unroll
  for (int m = 0; m < 8; ++m) {
    rmax[m] = -1e30f;
    rsum[m] = 0.f;
  }

  const int njt = ti + 1;  // causal: only j-tiles up to the diagonal
  for (int jt64 = 0; jt64 < njt; ++jt64) {
    const int j0 = jt64 * 64;
    const bool diag = (jt64 == ti);  // mask only needed on the diagonal tile
    {  // stage Q tile [j][d] via async b128 copies
      const int jr = tid >> 1, doff = (tid & 1) * 32;
      const __bf16* src = qb + (size_t)(b * kT + j0 + jr) * kC + h * kD + doff;
      __bf16* dst = &Qs[jr * 64 + doff];
      copy_b128_to_lds(src, dst);
      copy_b128_to_lds(src + 8, dst + 8);
      copy_b128_to_lds(src + 16, dst + 16);
      copy_b128_to_lds(src + 24, dst + 24);
    }
    {  // stage V tile transposed -> Vs[d][j] (scatter, VGPR path)
      const int jr = tid >> 1, doff = (tid & 1) * 32;
      const __bf16* src = vb + (size_t)(b * kT + j0 + jr) * kC + h * kD + doff;
#pragma unroll
      for (int e = 0; e < 32; ++e) Vs[(doff + e) * 64 + jr] = src[e];
    }
    wait_async_lds();
    __syncthreads();

    // S = K · Qᵀ : 16 x 64, contraction over d
    v8f s[4];
#pragma unroll
    for (int jt = 0; jt < 4; ++jt) {
      v16bf q0, q1;
      const __bf16* qp = &Qs[(jt * 16 + l15) * 64];
#pragma unroll
      for (int e = 0; e < 16; ++e) {
        q0[e] = qp[half * 16 + e];       // d slice 0..31
        q1[e] = qp[32 + half * 16 + e];  // d slice 32..63
      }
      v8f z = {};
      z = __builtin_amdgcn_wmma_f32_16x16x32_bf16(false, kf[0], false, q0, (short)0, z, false, false);
      s[jt] = __builtin_amdgcn_wmma_f32_16x16x32_bf16(false, kf[1], false, q1, (short)0, z, false, false);
    }

    // online softmax per row (row = m + half*8 in C-layout)
#pragma unroll
    for (int m = 0; m < 8; ++m) {
      const int gi = i0 + m + half * 8;
      float mx = -1e30f;
#pragma unroll
      for (int jt = 0; jt < 4; ++jt) {
        float v = s[jt][m] * 0.125f;  // 1/sqrt(64)
        if (diag) {
          const int gj = j0 + jt * 16 + l15;
          v = (gj <= gi) ? v : -1e30f;  // causal mask (diagonal tile only)
        }
        s[jt][m] = v;
        mx = fmaxf(mx, v);
      }
#pragma unroll
      for (int off = 1; off < 16; off <<= 1) mx = fmaxf(mx, __shfl_xor(mx, off, 32));
      const float mnew = fmaxf(rmax[m], mx);
      const float corr = __expf(rmax[m] - mnew);
      rsum[m] *= corr;
#pragma unroll
      for (int dt = 0; dt < 4; ++dt) o[dt][m] *= corr;
      float ls = 0.f;
#pragma unroll
      for (int jt = 0; jt < 4; ++jt) {
        const float p = __expf(s[jt][m] - mnew);
        s[jt][m] = p;
        ls += p;
      }
#pragma unroll
      for (int off = 1; off < 16; off <<= 1) ls += __shfl_xor(ls, off, 32);
      rsum[m] += ls;
      rmax[m] = mnew;
      // stash P (bf16) to per-wave LDS for the A-fragment reload
#pragma unroll
      for (int jt = 0; jt < 4; ++jt)
        Ps[wid][(m + half * 8) * 64 + jt * 16 + l15] = (__bf16)s[jt][m];
    }

    // O += P · V  (contraction over j)
    v16bf pf[2];
#pragma unroll
    for (int f = 0; f < 2; ++f) {
      const __bf16* pp = &Ps[wid][l15 * 64 + f * 32];
#pragma unroll
      for (int e = 0; e < 8; ++e) {
        pf[f][e] = pp[half * 16 + e];
        pf[f][8 + e] = pp[half * 16 + 8 + e];
      }
    }
#pragma unroll
    for (int dt = 0; dt < 4; ++dt) {
      v16bf v0, v1;
      const __bf16* vp = &Vs[(dt * 16 + l15) * 64];
#pragma unroll
      for (int e = 0; e < 16; ++e) {
        v0[e] = vp[half * 16 + e];       // j slice 0..31
        v1[e] = vp[32 + half * 16 + e];  // j slice 32..63
      }
      o[dt] = __builtin_amdgcn_wmma_f32_16x16x32_bf16(false, pf[0], false, v0, (short)0, o[dt], false, false);
      o[dt] = __builtin_amdgcn_wmma_f32_16x16x32_bf16(false, pf[1], false, v1, (short)0, o[dt], false, false);
    }
    __syncthreads();
  }

  // x1 = x + O / rowsum
#pragma unroll
  for (int dt = 0; dt < 4; ++dt)
#pragma unroll
    for (int m = 0; m < 8; ++m) {
      const int gi = i0 + m + half * 8;
      const int col = h * kD + dt * 16 + l15;
      const size_t idx = (size_t)(b * kT + gi) * kC + col;
      x1[idx] = x[idx] + o[dt][m] / rsum[m];
    }
}

// ---------------------------------------------------------------- launcher
extern "C" void kernel_launch(void* const* d_in, const int* in_sizes, int n_in,
                              void* d_out, int out_size, void* d_ws, size_t ws_size,
                              hipStream_t stream) {
  (void)in_sizes; (void)n_in; (void)out_size; (void)ws_size;
  const float* x    = (const float*)d_in[0];
  const float* ln1g = (const float*)d_in[1];
  const float* ln1b = (const float*)d_in[2];
  const float* Wq   = (const float*)d_in[3];
  const float* bq   = (const float*)d_in[4];
  const float* Wk   = (const float*)d_in[5];
  const float* bk   = (const float*)d_in[6];
  const float* Wv   = (const float*)d_in[7];
  const float* bv   = (const float*)d_in[8];
  const float* ln2g = (const float*)d_in[9];
  const float* ln2b = (const float*)d_in[10];
  const float* W1   = (const float*)d_in[11];
  const float* b1   = (const float*)d_in[12];
  const float* W2   = (const float*)d_in[13];
  const float* b2   = (const float*)d_in[14];
  float* out = (float*)d_out;

  char* w = (char*)d_ws;
  auto carve = [&](size_t bytes) -> char* {
    char* p = w;
    w += (bytes + 255) & ~(size_t)255;
    return p;
  };
  __bf16* hb   = (__bf16*)carve((size_t)kR * kC * 2);
  __bf16* h2b  = (__bf16*)carve((size_t)kR * kC * 2);
  __bf16* wqb  = (__bf16*)carve((size_t)kC * kC * 2);
  __bf16* wkb  = (__bf16*)carve((size_t)kC * kC * 2);
  __bf16* wvb  = (__bf16*)carve((size_t)kC * kC * 2);
  __bf16* w1b  = (__bf16*)carve((size_t)kC * kF * 2);
  __bf16* w2b  = (__bf16*)carve((size_t)kF * kC * 2);
  __bf16* qbuf = (__bf16*)carve((size_t)kR * kC * 2);
  __bf16* kbuf = (__bf16*)carve((size_t)kR * kC * 2);
  __bf16* vbuf = (__bf16*)carve((size_t)kR * kC * 2);
  float*  x1   = (float*)carve((size_t)kR * kC * 4);
  __bf16* gb   = (__bf16*)carve((size_t)kR * kF * 2);

  // one-time weight downcasts
  cast_f32_to_bf16<<<1024, 256, 0, stream>>>(Wq, wqb, kC * kC);
  cast_f32_to_bf16<<<1024, 256, 0, stream>>>(Wk, wkb, kC * kC);
  cast_f32_to_bf16<<<1024, 256, 0, stream>>>(Wv, wvb, kC * kC);
  cast_f32_to_bf16<<<2048, 256, 0, stream>>>(W1, w1b, kC * kF);
  cast_f32_to_bf16<<<2048, 256, 0, stream>>>(W2, w2b, kF * kC);

  // LN1
  layernorm_bf16<<<kR, 256, 0, stream>>>(x, ln1g, ln1b, hb);

  // QKV projections
  dim3 g1(kR / 128, kC / 128);
  gemm_bf16_wmma<0><<<g1, 256, 0, stream>>>(hb, wqb, bq, nullptr, qbuf, nullptr, kR, kC, kC);
  gemm_bf16_wmma<0><<<g1, 256, 0, stream>>>(hb, wkb, bk, nullptr, kbuf, nullptr, kR, kC, kC);
  gemm_bf16_wmma<0><<<g1, 256, 0, stream>>>(hb, wvb, bv, nullptr, vbuf, nullptr, kR, kC, kC);

  // causal attention + residual -> x1
  dim3 ga(kT / 64, kH, kB);
  attn_flash<<<ga, 128, 0, stream>>>(qbuf, kbuf, vbuf, x, x1);

  // LN2
  layernorm_bf16<<<kR, 256, 0, stream>>>(x1, ln2g, ln2b, h2b);

  // FFN
  dim3 g2(kR / 128, kF / 128);
  gemm_bf16_wmma<1><<<g2, 256, 0, stream>>>(h2b, w1b, b1, nullptr, gb, nullptr, kR, kF, kC);
  dim3 g3(kR / 128, kC / 128);
  gemm_bf16_wmma<2><<<g3, 256, 0, stream>>>(gb, w2b, b2, x1, nullptr, out, kR, kC, kF);
}
